// GCNHybrid_64484638982370
// MI455X (gfx1250) — compile-verified, matrix operand order checked by reading, not run on hardware
//
#include <hip/hip_runtime.h>

#define BB 2
#define TT 12
#define NN 10000
#define FF 16
#define HG 32
#define HH 64
#define PP 4
#define EE 160000
#define BT (BB*TT)
#define NTILES ((BB*NN)/16)     /* 1250 16-row tiles */
#define TPB_LSTM 128
#define NWAVE 4

typedef __attribute__((ext_vector_type(16))) __bf16 v16bf;
typedef __attribute__((ext_vector_type(8)))  float  v8f;

// ---- branch-free native activations (exp2/rcp are single trans ops) ----
__device__ __forceinline__ float fast_exp2(float x) { return __builtin_amdgcn_exp2f(x); }
__device__ __forceinline__ float fast_rcp(float x)  { return __builtin_amdgcn_rcpf(x); }

__device__ __forceinline__ float sigf(float x) {
  // 1/(1+e^-x); exp2(-inf)->0, exp2(+inf)->inf, rcp(inf)->0: correct saturation.
  return fast_rcp(1.0f + fast_exp2(-1.44269504088896340736f * x));
}
__device__ __forceinline__ float tanh_fast(float x) {
#if __has_builtin(__builtin_amdgcn_tanhf)
  return __builtin_amdgcn_tanhf(x);        // hardware V_TANH_F32 on gfx1250
#else
  return 2.0f * fast_rcp(1.0f + fast_exp2(-2.88539008177792681472f * x)) - 1.0f;
#endif
}

// Assemble a 16x32 bf16 WMMA A/B operand from a row-major LDS tile
// (row stride in elements, must be multiple of 8; 16B aligned).
// Layout per ISA: lanes 0-15 hold K 0-7 & 16-23, lanes 16-31 hold K 8-15 & 24-31.
__device__ __forceinline__ v16bf lds_ab(const __bf16* base, int stride, int lane) {
  const __bf16* p = base + (lane & 15) * stride + ((lane >> 4) << 3);
  union { v16bf v; int4 q[2]; } u;
  u.q[0] = *(const int4*)(p);
  u.q[1] = *(const int4*)(p + 16);
  return u.v;
}

// Assemble a 16x32 A operand from row-major f32 global memory (row stride 32 floats),
// converting to bf16. p must already point at (row, koff) for this lane.
__device__ __forceinline__ v16bf glb_a_f32(const float* p) {
  float4 q0 = *(const float4*)(p + 0);
  float4 q1 = *(const float4*)(p + 4);
  float4 q2 = *(const float4*)(p + 16);
  float4 q3 = *(const float4*)(p + 20);
  v16bf v;
  v[0]  = (__bf16)q0.x; v[1]  = (__bf16)q0.y; v[2]  = (__bf16)q0.z; v[3]  = (__bf16)q0.w;
  v[4]  = (__bf16)q1.x; v[5]  = (__bf16)q1.y; v[6]  = (__bf16)q1.z; v[7]  = (__bf16)q1.w;
  v[8]  = (__bf16)q2.x; v[9]  = (__bf16)q2.y; v[10] = (__bf16)q2.z; v[11] = (__bf16)q2.w;
  v[12] = (__bf16)q3.x; v[13] = (__bf16)q3.y; v[14] = (__bf16)q3.z; v[15] = (__bf16)q3.w;
  return v;
}

#define WMMA_BF16(A, B, C) \
  __builtin_amdgcn_wmma_f32_16x16x32_bf16(false, (A), false, (B), (short)0, (C), false, false)

// ---------------- GCN kernels ----------------

__global__ void k_zero_f32(float* p, int n) {
  int i = blockIdx.x * 256 + threadIdx.x;
  if (i < n) p[i] = 0.0f;
}

__global__ void k_degree(const int* __restrict__ ei, float* __restrict__ deg) {
  int e = blockIdx.x * 256 + threadIdx.x;
  if (e < EE) atomicAdd(deg + ei[EE + e], 1.0f);
}

__global__ void k_dinv(float* __restrict__ deg) {
  int n = blockIdx.x * 256 + threadIdx.x;
  if (n < NN) deg[n] = rsqrtf(deg[n] + 1.0f);   // +1 for self loop
}

// xw = x @ gcn_w ; out = gcn_b + dinv[n]^2 * xw  (self-loop message pre-added)
__global__ void k_gcn_xw(const float* __restrict__ x, const float* __restrict__ wmat,
                         const float* __restrict__ bvec, const float* __restrict__ dinv,
                         float* __restrict__ xw, float* __restrict__ outb) {
  __shared__ float sw[FF * HG];
  int tid = threadIdx.x;
  for (int i = tid; i < FF * HG; i += 256) sw[i] = wmat[i];
  __syncthreads();
  size_t id = (size_t)blockIdx.x * 256 + tid;
  if (id >= (size_t)BT * NN * HG) return;
  int c = (int)(id & (HG - 1));
  size_t row = id >> 5;
  int n = (int)(row % NN);
  const float* xr = x + row * FF;
  float acc = 0.0f;
#pragma unroll
  for (int k = 0; k < FF; ++k) acc += xr[k] * sw[k * HG + c];
  float dn = dinv[n];
  xw[id]   = acc;
  outb[id] = bvec[c] + acc * dn * dn;
}

// one edge per 32 lanes; lane = channel
__global__ void k_gcn_scatter(const int* __restrict__ ei, const float* __restrict__ dinv,
                              const float* __restrict__ xw, float* __restrict__ outb) {
  int bt = blockIdx.y;
  int e = blockIdx.x * 8 + (threadIdx.x >> 5);
  if (e >= EE) return;
  int c = threadIdx.x & 31;
  int s = ei[e], d = ei[EE + e];
  float nrm = dinv[s] * dinv[d];
  float v = xw[((size_t)bt * NN + s) * HG + c] * nrm;
  atomicAdd(outb + ((size_t)bt * NN + d) * HG + c, v);
}

// ---------------- fused LSTM1 + LSTM2 + FC ----------------
// One wave handles 16 rows (m = b*N + n0..n0+15). Gates[16x256] computed as
// 16 WMMA N-tiles; tile nt = gate*4 + jt so i/f/g/o for an element share (v,lane).

__global__ __launch_bounds__(TPB_LSTM) void k_lstm_fused(
    const float* __restrict__ gout, const float* __restrict__ zeta,
    const float* __restrict__ w_ih1, const float* __restrict__ w_hh1,
    const float* __restrict__ b_ih1, const float* __restrict__ b_hh1,
    const float* __restrict__ w_ih2, const float* __restrict__ w_hh2,
    const float* __restrict__ b_ih2, const float* __restrict__ b_hh2,
    const float* __restrict__ fc_w, const float* __restrict__ fc_b,
    float* __restrict__ out) {
  __shared__ __align__(16) __bf16 s_wih1[256 * 40];     // [g][k] pad->40
  __shared__ __align__(16) __bf16 s_whh1[256 * 72];     // [g][k] pad->72
  __shared__ __align__(16) __bf16 s_whh2[256 * 72];
  __shared__ float s_b1[256], s_b2[256], s_wi2[256];
  __shared__ __align__(16) __bf16 s_h[NWAVE][2][16 * 72];  // h ping-pong, A layout
  __shared__ float s_z[NWAVE][TT * 16];
  __shared__ float s_h1f[NWAVE][16 * 68];
  __shared__ float s_h2f[NWAVE][16 * 68];

  const int tid = threadIdx.x;
  for (int i = tid; i < 256 * 32; i += TPB_LSTM) s_wih1[(i >> 5) * 40 + (i & 31)] = (__bf16)w_ih1[i];
  for (int i = tid; i < 256 * 64; i += TPB_LSTM) s_whh1[(i >> 6) * 72 + (i & 63)] = (__bf16)w_hh1[i];
  for (int i = tid; i < 256 * 64; i += TPB_LSTM) s_whh2[(i >> 6) * 72 + (i & 63)] = (__bf16)w_hh2[i];
  for (int i = tid; i < 256; i += TPB_LSTM) {
    s_b1[i]  = b_ih1[i] + b_hh1[i];
    s_b2[i]  = b_ih2[i] + b_hh2[i];
    s_wi2[i] = w_ih2[i];
  }
  __syncthreads();

  const int w    = tid >> 5;
  const int lane = tid & 31;
  const int lo   = lane & 15;
  const int hi   = lane >> 4;
  const int tile = blockIdx.x * NWAVE + w;
  if (tile >= NTILES) return;
  const int b  = tile / (NN / 16);
  const int n0 = (tile - b * (NN / 16)) * 16;

  // zeta for this tile: s_z[t][m]
  for (int i = lane; i < TT * 16; i += 32) {
    int t = i >> 4, m = i & 15;
    s_z[w][i] = zeta[((size_t)b * TT + t) * NN + (n0 + m)];
  }

  __bf16* hc = &s_h[w][0][0];
  __bf16* hn = &s_h[w][1][0];
  for (int i = lane; i < 16 * 72; i += 32) hc[i] = (__bf16)0.0f;

  float cr[32];
#pragma unroll
  for (int i = 0; i < 32; ++i) cr[i] = 0.0f;

  // ---------- LSTM 1 ----------
  for (int t = 0; t < TT; ++t) {
    const float* xrow = gout + (((size_t)(b * TT + t)) * NN + n0) * HG;
    v16bf ax  = glb_a_f32(xrow + (size_t)lo * HG + (hi << 3));
    v16bf ah0 = lds_ab(hc, 72, lane);
    v16bf ah1 = lds_ab(hc + 32, 72, lane);
#pragma unroll
    for (int jt = 0; jt < 4; ++jt) {
      v8f G[4];
#pragma unroll
      for (int gg = 0; gg < 4; ++gg) {
        int nt = gg * 4 + jt;
        float bv = s_b1[nt * 16 + lo];
        v8f acc = {bv, bv, bv, bv, bv, bv, bv, bv};
        acc = WMMA_BF16(ax,  lds_ab(&s_wih1[nt * 16 * 40], 40, lane), acc);
        acc = WMMA_BF16(ah0, lds_ab(&s_whh1[nt * 16 * 72], 72, lane), acc);
        acc = WMMA_BF16(ah1, lds_ab(&s_whh1[nt * 16 * 72 + 32], 72, lane), acc);
        G[gg] = acc;
      }
#pragma unroll
      for (int v = 0; v < 8; ++v) {
        float iv = sigf(G[0][v]);
        float fv = sigf(G[1][v]);
        float gv = tanh_fast(G[2][v]);
        float ov = sigf(G[3][v]);
        float cv = fv * cr[jt * 8 + v] + iv * gv;
        cr[jt * 8 + v] = cv;
        float hv = ov * tanh_fast(cv);
        int m = v + 8 * hi;
        hn[m * 72 + jt * 16 + lo] = (__bf16)hv;
        if (t == TT - 1) s_h1f[w][m * 68 + jt * 16 + lo] = hv;
      }
    }
    __bf16* tp = hc; hc = hn; hn = tp;
  }

  // ---------- LSTM 2 (input dim 1: xi = z*w_ih2 + biases, folded into C init) ----------
  for (int i = lane; i < 16 * 72; i += 32) hc[i] = (__bf16)0.0f;
#pragma unroll
  for (int i = 0; i < 32; ++i) cr[i] = 0.0f;

  for (int t = 0; t < TT; ++t) {
    v16bf ah0 = lds_ab(hc, 72, lane);
    v16bf ah1 = lds_ab(hc + 32, 72, lane);
    float zr[8];
#pragma unroll
    for (int v = 0; v < 8; ++v) zr[v] = s_z[w][t * 16 + v + 8 * hi];
#pragma unroll
    for (int jt = 0; jt < 4; ++jt) {
      v8f G[4];
#pragma unroll
      for (int gg = 0; gg < 4; ++gg) {
        int nt = gg * 4 + jt;
        float bv = s_b2[nt * 16 + lo];
        float wv = s_wi2[nt * 16 + lo];
        v8f acc;
#pragma unroll
        for (int v = 0; v < 8; ++v) acc[v] = bv + zr[v] * wv;
        acc = WMMA_BF16(ah0, lds_ab(&s_whh2[nt * 16 * 72], 72, lane), acc);
        acc = WMMA_BF16(ah1, lds_ab(&s_whh2[nt * 16 * 72 + 32], 72, lane), acc);
        G[gg] = acc;
      }
#pragma unroll
      for (int v = 0; v < 8; ++v) {
        float iv = sigf(G[0][v]);
        float fv = sigf(G[1][v]);
        float gv = tanh_fast(G[2][v]);
        float ov = sigf(G[3][v]);
        float cv = fv * cr[jt * 8 + v] + iv * gv;
        cr[jt * 8 + v] = cv;
        float hv = ov * tanh_fast(cv);
        int m = v + 8 * hi;
        hn[m * 72 + jt * 16 + lo] = (__bf16)hv;
        if (t == TT - 1) s_h2f[w][m * 68 + jt * 16 + lo] = hv;
      }
    }
    __bf16* tp = hc; hc = hn; hn = tp;
  }

  // ---------- FC: y[m,p] = [h1 h2] . fc_w[p,:] + fc_b[p] ----------
  const float* h1 = s_h1f[w];
  const float* h2 = s_h2f[w];
  for (int p = hi; p < PP; p += 2) {
    float acc = fc_b[p];
#pragma unroll 8
    for (int k = 0; k < HH; ++k) acc += h1[lo * 68 + k] * fc_w[p * (2 * HH) + k];
#pragma unroll 8
    for (int k = 0; k < HH; ++k) acc += h2[lo * 68 + k] * fc_w[p * (2 * HH) + HH + k];
    out[((size_t)b * PP + p) * NN + (n0 + lo)] = acc;
  }
}

extern "C" void kernel_launch(void* const* d_in, const int* in_sizes, int n_in,
                              void* d_out, int out_size, void* d_ws, size_t ws_size,
                              hipStream_t stream) {
  (void)in_sizes; (void)n_in; (void)out_size; (void)ws_size;
  const float* era5  = (const float*)d_in[0];
  const float* zeta  = (const float*)d_in[1];
  const int*   ei    = (const int*)d_in[2];
  const float* gcn_w = (const float*)d_in[3];
  const float* gcn_b = (const float*)d_in[4];
  const float* w_ih1 = (const float*)d_in[5];
  const float* w_hh1 = (const float*)d_in[6];
  const float* b_ih1 = (const float*)d_in[7];
  const float* b_hh1 = (const float*)d_in[8];
  const float* w_ih2 = (const float*)d_in[9];
  const float* w_hh2 = (const float*)d_in[10];
  const float* b_ih2 = (const float*)d_in[11];
  const float* b_hh2 = (const float*)d_in[12];
  const float* fc_w  = (const float*)d_in[13];
  const float* fc_b  = (const float*)d_in[14];
  float* out = (float*)d_out;

  // workspace layout
  float* dinv = (float*)d_ws;                                   // N f32 (deg -> dinv in place)
  float* xw   = (float*)((char*)d_ws + 65536);                  // BT*N*HG f32
  float* gout = xw + (size_t)BT * NN * HG;                      // BT*N*HG f32

  k_zero_f32<<<(NN + 255) / 256, 256, 0, stream>>>(dinv, NN);
  k_degree<<<(EE + 255) / 256, 256, 0, stream>>>(ei, dinv);
  k_dinv<<<(NN + 255) / 256, 256, 0, stream>>>(dinv);

  size_t tot = (size_t)BT * NN * HG;
  k_gcn_xw<<<(unsigned)((tot + 255) / 256), 256, 0, stream>>>(era5, gcn_w, gcn_b, dinv, xw, gout);

  dim3 gsc((EE + 7) / 8, BT);
  k_gcn_scatter<<<gsc, 256, 0, stream>>>(ei, dinv, xw, gout);

  int nblk = (NTILES + NWAVE - 1) / NWAVE;   // 313
  k_lstm_fused<<<nblk, TPB_LSTM, 0, stream>>>(gout, zeta,
                                              w_ih1, w_hh1, b_ih1, b_hh1,
                                              w_ih2, w_hh2, b_ih2, b_hh2,
                                              fc_w, fc_b, out);
}
